// Transformer_28140625723594
// MI455X (gfx1250) — matrix operand order repaired and checked by reading, not compiled, not run on hardware
//
#include <hip/hip_runtime.h>
#include <hip/hip_bf16.h>

typedef __attribute__((ext_vector_type(8)))  _Float16 v8h;
typedef __attribute__((ext_vector_type(16))) _Float16 v16h;
typedef __attribute__((ext_vector_type(8)))  float    v8f;

#define BB   2
#define SS   2048
#define DDIM 1024
#define HH   16
#define HDD  64
// fold 1/sqrt(64) * log2(e) into stored Q so softmax uses exp2 directly
#define QSCALE 0.1803368801111137f

__device__ __forceinline__ v16h cat8(v8h lo, v8h hi) {
  return __builtin_shufflevector(lo, hi, 0,1,2,3,4,5,6,7,8,9,10,11,12,13,14,15);
}
__device__ __forceinline__ v16h ld16(const _Float16* p) {
  return cat8(*(const v8h*)p, *(const v8h*)(p + 8));
}
__device__ __forceinline__ v8f wmma16(v16h a, v16h b, v8f c) {
  return __builtin_amdgcn_wmma_f32_16x16x32_f16(false, a, false, b, (short)0, c, false, false);
}

// ---------------------------------------------------------------- convert
__global__ __launch_bounds__(256) void cvt_kernel(
    const float* __restrict__ x,  const float* __restrict__ w0,
    const float* __restrict__ w1, const float* __restrict__ w2,
    _Float16* __restrict__ xh, _Float16* __restrict__ wh) {
  const int nx = BB * SS * DDIM;     // 4,194,304
  const int nw = DDIM * DDIM;        // 1,048,576
  int stride = gridDim.x * blockDim.x;
  for (int i = blockIdx.x * blockDim.x + threadIdx.x; i < nx + 3 * nw; i += stride) {
    if (i < nx) {
      xh[i] = (_Float16)x[i];
    } else {
      int j = i - nx;
      float v = (j < nw) ? w0[j] : (j < 2 * nw) ? w1[j - nw] : w2[j - 2 * nw];
      wh[j] = (_Float16)v;
    }
  }
}

// ------------------------------------------------------------- QKV GEMM
// y[m,n] = sum_k x[m,k] * W[n,k] + b[n]   (m over 4096, n over 1024)
// Wave tile: 32(M) x 64(N), software-pipelined; block = 8 waves = 128x128.
__global__ __launch_bounds__(256) void qkv_gemm_kernel(
    const _Float16* __restrict__ xh, const _Float16* __restrict__ wh,
    const float* __restrict__ bq, const float* __restrict__ bk,
    const float* __restrict__ bv,
    _Float16* __restrict__ qh, _Float16* __restrict__ kh,
    _Float16* __restrict__ vh) {
  const int z = blockIdx.z;                       // 0=Q 1=K 2=V
  const _Float16* w  = wh + (size_t)z * DDIM * DDIM;
  const float* bias  = (z == 0) ? bq : (z == 1) ? bk : bv;
  _Float16* out      = (z == 0) ? qh : (z == 1) ? kh : vh;

  const int lane = threadIdx.x & 31;
  const int wave = threadIdx.x >> 5;
  const int g    = lane >> 4;        // half-wave group
  const int nlo  = lane & 15;

  const int m0 = blockIdx.x * 128 + (wave >> 1) * 32;
  const int n0 = blockIdx.y * 128 + (wave & 1) * 64;

  const _Float16* arow0 = xh + (size_t)(m0 + nlo) * DDIM;  // A rows m = lane&15
  const _Float16* arow1 = arow0 + (size_t)16 * DDIM;
  const _Float16* brow  = w + (size_t)(n0 + nlo) * DDIM;   // B col n = lane&15

  // prologue: k-step 0
  v16h a0 = cat8(*(const v8h*)(arow0 + 16 * g), *(const v8h*)(arow0 + 8 + 16 * g));
  v16h a1 = cat8(*(const v8h*)(arow1 + 16 * g), *(const v8h*)(arow1 + 8 + 16 * g));
  v16h b0 = ld16(brow + 16 * g);
  v16h b1 = ld16(brow + (size_t)16 * DDIM + 16 * g);
  v16h b2 = ld16(brow + (size_t)32 * DDIM + 16 * g);
  v16h b3 = ld16(brow + (size_t)48 * DDIM + 16 * g);

  v8f acc[8] = {};
#pragma unroll 2
  for (int kk = 0; kk < DDIM; kk += 32) {
    // prefetch next k-step into distinct regs (dummy kk=0 on last iter)
    const int kn = (kk + 32 < DDIM) ? (kk + 32) : 0;
    v16h na0 = cat8(*(const v8h*)(arow0 + kn + 16 * g),
                    *(const v8h*)(arow0 + kn + 8 + 16 * g));
    v16h na1 = cat8(*(const v8h*)(arow1 + kn + 16 * g),
                    *(const v8h*)(arow1 + kn + 8 + 16 * g));
    v16h nb0 = ld16(brow + kn + 16 * g);
    v16h nb1 = ld16(brow + (size_t)16 * DDIM + kn + 16 * g);
    v16h nb2 = ld16(brow + (size_t)32 * DDIM + kn + 16 * g);
    v16h nb3 = ld16(brow + (size_t)48 * DDIM + kn + 16 * g);

    acc[0] = wmma16(a0, b0, acc[0]);
    acc[1] = wmma16(a0, b1, acc[1]);
    acc[2] = wmma16(a0, b2, acc[2]);
    acc[3] = wmma16(a0, b3, acc[3]);
    acc[4] = wmma16(a1, b0, acc[4]);
    acc[5] = wmma16(a1, b1, acc[5]);
    acc[6] = wmma16(a1, b2, acc[6]);
    acc[7] = wmma16(a1, b3, acc[7]);

    a0 = na0; a1 = na1; b0 = nb0; b1 = nb1; b2 = nb2; b3 = nb3;
  }

  const float scale = (z == 0) ? QSCALE : 1.0f;
#pragma unroll
  for (int ms = 0; ms < 2; ++ms) {
#pragma unroll
    for (int t = 0; t < 4; ++t) {
      const v8f a = acc[ms * 4 + t];
      const int n = n0 + t * 16 + nlo;
      const float bn = bias[n];
      const int h = n >> 6, hd = n & 63;
#pragma unroll
      for (int v = 0; v < 8; ++v) {
        const int m = m0 + ms * 16 + 8 * g + v;
        const int b = m >> 11, s = m & (SS - 1);
        const float val = (a[v] + bn) * scale;
        out[(((size_t)(b * HH + h) * SS + s) * HDD) + hd] = (_Float16)val;
      }
    }
  }
}

// --------------------------------------------------------- flash attention
// grid: (S/64, B*H); block: 128 (4 waves). Wave owns 16 q rows.
// Scores computed transposed (scores^T = K * Q^T) so each lane owns one q
// column: softmax reductions are in-lane + a single shfl_xor(16).
__global__ __launch_bounds__(128) void attn_kernel(
    const _Float16* __restrict__ qh, const _Float16* __restrict__ kh,
    const _Float16* __restrict__ vh, float* __restrict__ out) {
  __shared__ __align__(16) _Float16 Klds[32][72];      // [key][d], padded
  __shared__ __align__(16) _Float16 Vt[64][40];        // [d][key], padded
  __shared__ __align__(16) _Float16 Pbuf[4][16][40];   // per-wave P[q][key]

  const int bh  = blockIdx.y;              // b*H + h
  const int q0  = blockIdx.x * 64;
  const int tid = threadIdx.x;
  const int lane = tid & 31, w = tid >> 5;
  const int g = lane >> 4, nlo = lane & 15;
  const int qbase = q0 + w * 16;

  const _Float16* qhead = qh + (size_t)bh * SS * HDD;
  const _Float16* khead = kh + (size_t)bh * SS * HDD;
  const _Float16* vhead = vh + (size_t)bh * SS * HDD;

  // preload Q^T as B-layout tiles: column n = q, contraction = d
  const _Float16* qrow = qhead + (size_t)(qbase + nlo) * HDD;
  v16h bQ0 = cat8(*(const v8h*)(qrow + 8 * g), *(const v8h*)(qrow + 16 + 8 * g));        // d in [0,32)
  v16h bQ1 = cat8(*(const v8h*)(qrow + 32 + 8 * g), *(const v8h*)(qrow + 48 + 8 * g));   // d in [32,64)

  float mcur = -INFINITY, lcur = 0.0f;
  v8f o0 = {}, o1 = {}, o2 = {}, o3 = {};

  const int ntiles = (q0 + 64) >> 5;     // causal upper bound for this block
  for (int jt = 0; jt < ntiles; ++jt) {
    const int jn = jt << 5;
    __syncthreads();
    // ---- cooperative staging: K row-major, V transposed
#pragma unroll
    for (int c = tid; c < 256; c += 128) {
      const int r = c >> 3, cc = (c & 7) * 8;
      *(v8h*)&Klds[r][cc] = *(const v8h*)(khead + (size_t)(jn + r) * HDD + cc);
    }
    {
      const int r = tid & 31, cb = (tid >> 5) * 16;
      const _Float16* vp = vhead + (size_t)(jn + r) * HDD + cb;
      v8h va = *(const v8h*)vp, vb = *(const v8h*)(vp + 8);
#pragma unroll
      for (int i = 0; i < 8; ++i) Vt[cb + i][r] = va[i];
#pragma unroll
      for (int i = 0; i < 8; ++i) Vt[cb + 8 + i][r] = vb[i];
    }
    if (jt + 1 < ntiles) {               // gfx1250 global_prefetch for next tile
      const size_t nr = (size_t)(jn + 32 + (tid & 31)) * HDD;
      __builtin_prefetch(khead + nr, 0, 1);
      __builtin_prefetch(vhead + nr, 0, 1);
    }
    __syncthreads();

    if (jn > qbase + 15) continue;       // wave-uniform: EXEC stays all-1s

    // ---- scores^T = K @ Q^T : A = K tiles (rows = keys), B = Q^T
    v16h aK00 = cat8(*(const v8h*)&Klds[nlo][8 * g],
                     *(const v8h*)&Klds[nlo][16 + 8 * g]);
    v16h aK01 = cat8(*(const v8h*)&Klds[nlo][32 + 8 * g],
                     *(const v8h*)&Klds[nlo][48 + 8 * g]);
    v16h aK10 = cat8(*(const v8h*)&Klds[16 + nlo][8 * g],
                     *(const v8h*)&Klds[16 + nlo][16 + 8 * g]);
    v16h aK11 = cat8(*(const v8h*)&Klds[16 + nlo][32 + 8 * g],
                     *(const v8h*)&Klds[16 + nlo][48 + 8 * g]);
    v8f s0 = {}, s1 = {};                // s0: keys jn+0..15, s1: jn+16..31
    s0 = wmma16(aK00, bQ0, s0);
    s1 = wmma16(aK10, bQ0, s1);
    s0 = wmma16(aK01, bQ1, s0);
    s1 = wmma16(aK11, bQ1, s1);

    // ---- causal mask + online softmax (lane owns q column = qbase+nlo)
    const int qcol = qbase + nlo;
    float tmax = -3.0e38f;
#pragma unroll
    for (int v = 0; v < 8; ++v) {
      const int key0 = jn + 8 * g + v;
      const float a = (key0 <= qcol) ? s0[v] : -3.0e38f;
      const float b = (key0 + 16 <= qcol) ? s1[v] : -3.0e38f;
      s0[v] = a; s1[v] = b;
      tmax = fmaxf(tmax, fmaxf(a, b));
    }
    tmax = fmaxf(tmax, __shfl_xor(tmax, 16, 32));     // combine key halves
    const float mn = fmaxf(mcur, tmax);
    const float alpha = __builtin_amdgcn_exp2f(mcur - mn);
    mcur = mn;
    float rs = 0.0f;
#pragma unroll
    for (int v = 0; v < 8; ++v) {
      const float p0 = __builtin_amdgcn_exp2f(s0[v] - mn);
      const float p1 = __builtin_amdgcn_exp2f(s1[v] - mn);
      rs += p0 + p1;
      Pbuf[w][nlo][8 * g + v]      = (_Float16)p0;   // store transposed: P[q][k]
      Pbuf[w][nlo][16 + 8 * g + v] = (_Float16)p1;
    }
    rs += __shfl_xor(rs, 16, 32);
    lcur = lcur * alpha + rs;

    // broadcast alpha to this lane's output rows q = qbase + 8g+v
    float av[8];
#pragma unroll
    for (int v = 0; v < 8; ++v) av[v] = __shfl(alpha, 8 * g + v, 16);
#pragma unroll
    for (int v = 0; v < 8; ++v) {
      o0[v] *= av[v]; o1[v] *= av[v]; o2[v] *= av[v]; o3[v] *= av[v];
    }

    // same-wave LDS RAW: DS ops are in-order per wave; make the wait explicit
    asm volatile("s_wait_dscnt 0" ::: "memory");

    // ---- P @ V : A = P (A-layout from LDS), B = V^T tiles
    v16h aP  = cat8(*(const v8h*)&Pbuf[w][nlo][16 * g],
                    *(const v8h*)&Pbuf[w][nlo][8 + 16 * g]);
    v16h bV0 = ld16(&Vt[nlo][16 * g]);
    v16h bV1 = ld16(&Vt[16 + nlo][16 * g]);
    v16h bV2 = ld16(&Vt[32 + nlo][16 * g]);
    v16h bV3 = ld16(&Vt[48 + nlo][16 * g]);
    o0 = wmma16(aP, bV0, o0);
    o1 = wmma16(aP, bV1, o1);
    o2 = wmma16(aP, bV2, o2);
    o3 = wmma16(aP, bV3, o3);
  }

  // ---- normalize (broadcast 1/l to output rows) and store fp32 [B,S,D]
  const float inv = 1.0f / lcur;
  float iv[8];
#pragma unroll
  for (int v = 0; v < 8; ++v) iv[v] = __shfl(inv, 8 * g + v, 16);
  const int b = bh >> 4, h = bh & 15;
#pragma unroll
  for (int v = 0; v < 8; ++v) {
    const int q = qbase + 8 * g + v;
    float* orow = out + ((size_t)b * SS + q) * DDIM + h * HDD;
    orow[nlo]      = o0[v] * iv[v];
    orow[16 + nlo] = o1[v] * iv[v];
    orow[32 + nlo] = o2[v] * iv[v];
    orow[48 + nlo] = o3[v] * iv[v];
  }
}

// ---------------------------------------------------------------- launch
extern "C" void kernel_launch(void* const* d_in, const int* in_sizes, int n_in,
                              void* d_out, int out_size, void* d_ws, size_t ws_size,
                              hipStream_t stream) {
  (void)in_sizes; (void)n_in; (void)out_size; (void)ws_size;
  const float* x  = (const float*)d_in[0];
  const float* wq = (const float*)d_in[1];
  const float* bq = (const float*)d_in[2];
  const float* wk = (const float*)d_in[3];
  const float* bk = (const float*)d_in[4];
  const float* wv = (const float*)d_in[5];
  const float* bv = (const float*)d_in[6];
  float* out = (float*)d_out;

  _Float16* xh = (_Float16*)d_ws;                       // 4096*1024 halfs
  _Float16* wh = xh + (size_t)BB * SS * DDIM;           // 3*1024*1024 halfs
  _Float16* qh = wh + (size_t)3 * DDIM * DDIM;          // [B*H, S, 64]
  _Float16* kh = qh + (size_t)BB * SS * DDIM;
  _Float16* vh = kh + (size_t)BB * SS * DDIM;

  cvt_kernel<<<1024, 256, 0, stream>>>(x, wq, wk, wv, xh, wh);
  qkv_gemm_kernel<<<dim3(32, 8, 3), 256, 0, stream>>>(xh, wh, bq, bk, bv, qh, kh, vh);
  attn_kernel<<<dim3(SS / 64, BB * HH), 128, 0, stream>>>(qh, kh, vh, out);
}
